// GraphDecoderHet_16432544874898
// MI455X (gfx1250) — matrix operand order repaired and checked by reading, compiled, not run on hardware
//
#include <hip/hip_runtime.h>
#include <hip/hip_bf16.h>

typedef __attribute__((ext_vector_type(16))) _Float16 v16h;
typedef __attribute__((ext_vector_type(4)))  _Float16 v4h;
typedef __attribute__((ext_vector_type(8)))  float    v8f;

#define R_REL 8

static __device__ __forceinline__ void atomAddF32(float* p, float v) {
  __hip_atomic_fetch_add(p, v, __ATOMIC_RELAXED, __HIP_MEMORY_SCOPE_AGENT);
}

// ---------------- graph prep ----------------
__global__ __launch_bounds__(256) void k_count(const int* __restrict__ dst,
                                               const int* __restrict__ et,
                                               float* __restrict__ cnt, int E) {
  int t = blockIdx.x * 256 + threadIdx.x;
  if (t >= E) return;
  atomAddF32(&cnt[(size_t)dst[t] * R_REL + et[t]], 1.0f);
}

__global__ __launch_bounds__(256) void k_scale_ew(const int* __restrict__ dst,
                                                  const int* __restrict__ et,
                                                  const float* __restrict__ ew,
                                                  const float* __restrict__ cnt,
                                                  float* __restrict__ ew2, int E) {
  int t = blockIdx.x * 256 + threadIdx.x;
  if (t >= E) return;
  float c = cnt[(size_t)dst[t] * R_REL + et[t]];
  ew2[t] = ew[t] / fmaxf(c, 1.0f);
}

// ---------------- conversions ----------------
__global__ __launch_bounds__(256) void k_f32_to_f16(const float* __restrict__ x,
                                                    _Float16* __restrict__ xh, int n) {
  int t = blockIdx.x * 256 + threadIdx.x;
  if (t < n) xh[t] = (_Float16)x[t];
}

// Pre-swizzle W[r][k][n] (f32 row-major) into per-lane WMMA B fragments (f16).
// Fragment element order: (((r*KT+kt)*NT+nt)*32 + lane)*16 + i
// B layout (16-bit, 32x16): lanes 0-15 hold K=0..7/16..23 (pairs per VGPR),
// lanes 16-31 hold K=8..15/24..31; column n = lane&15.
template<int DIN, int DOUT>
__global__ __launch_bounds__(256) void k_prep_b(const float* __restrict__ W,
                                                _Float16* __restrict__ bswz) {
  constexpr int KT = DIN / 32;
  constexpr int NT = DOUT / 16;
  int t = blockIdx.x * 256 + threadIdx.x;
  if (t >= R_REL * KT * NT * 32 * 16) return;
  int i    = t & 15;
  int lane = (t >> 4) & 31;
  int rest = t >> 9;
  int nt = rest % NT; rest /= NT;
  int kt = rest % KT;
  int r  = rest / KT;
  int k = kt * 32 + ((lane >> 4) * 8) + (i & 7) + ((i >> 3) * 16);
  int n = nt * 16 + (lane & 15);
  bswz[t] = (_Float16)W[((size_t)r * DIN + k) * DOUT + n];
}

// ---------------- WMMA GEMM: z[r] = xh @ W[r] ----------------
// One wave per (16-row strip, relation). A fragments held in VGPRs across
// all k-tiles; B fragments double-buffered from the pre-swizzled buffer.
template<int DIN, int DOUT>
__global__ __launch_bounds__(32) void k_gemm(const _Float16* __restrict__ xh,
                                             const _Float16* __restrict__ bswz,
                                             _Float16* __restrict__ zh, int N) {
  constexpr int KT = DIN / 32;
  constexpr int NT = DOUT / 16;
  const int mt   = blockIdx.x;
  const int r    = blockIdx.y;
  const int lane = threadIdx.x;

  int m = mt * 16 + (lane & 15);
  if (m >= N) m = N - 1;                 // N is a multiple of 16 in practice
  const int kb = (lane >> 4) * 8;        // A layout: lanes 0-15 K 0-7/16-23, 16-31 K 8-15/24-31

  union { v16h h; uint4 u[2]; } a[KT];
  const _Float16* xrow = xh + (size_t)m * DIN;
#pragma unroll
  for (int kt = 0; kt < KT; ++kt) {
    a[kt].u[0] = *(const uint4*)(xrow + kt * 32 + kb);
    a[kt].u[1] = *(const uint4*)(xrow + kt * 32 + kb + 16);
  }

  const _Float16* bb = bswz + (size_t)r * KT * NT * 32 * 16;
  _Float16* zt = zh + ((size_t)r * N + (size_t)mt * 16) * DOUT;
  const int lrow = (lane >> 4) * 8;      // C/D layout: VGPR j -> row j (+8 for hi lanes)
  const int lcol = lane & 15;
  const bool fullTile = (mt * 16 + 16 <= N);   // uniform scalar guard

#pragma unroll
  for (int nt = 0; nt < NT; ++nt) {
    if (nt + 1 < NT)
      __builtin_prefetch(bb + ((size_t)(nt + 1) * 32 + lane) * 16, 0, 1);
    v8f c = {};
    v16h bcur = *(const v16h*)(bb + ((size_t)nt * 32 + lane) * 16);
#pragma unroll
    for (int kt = 0; kt < KT; ++kt) {
      v16h bnext;
      if (kt + 1 < KT)
        bnext = *(const v16h*)(bb + ((size_t)((kt + 1) * NT + nt) * 32 + lane) * 16);
      c = __builtin_amdgcn_wmma_f32_16x16x32_f16(false, a[kt].h, false, bcur,
                                                 (short)0, c, false, false);
      if (kt + 1 < KT) bcur = bnext;
    }
    if (fullTile) {
#pragma unroll
      for (int j = 0; j < 8; ++j)
        zt[(size_t)(lrow + j) * DOUT + nt * 16 + lcol] = (_Float16)c[j];
    } else {
#pragma unroll
      for (int j = 0; j < 8; ++j)
        if (mt * 16 + lrow + j < N)
          zt[(size_t)(lrow + j) * DOUT + nt * 16 + lcol] = (_Float16)c[j];
    }
  }
}

// ---------------- edge scatter: acc[dst] += ew' * z[etype][src] ----------------
template<int DOUT>
__global__ __launch_bounds__(256) void k_edge_agg(const int* __restrict__ src,
                                                  const int* __restrict__ dst,
                                                  const int* __restrict__ et,
                                                  const float* __restrict__ ew2,
                                                  const _Float16* __restrict__ zh,
                                                  float* __restrict__ acc,
                                                  int E, int N) {
  constexpr int CG = DOUT / 4;           // 4 columns per thread
  unsigned t = blockIdx.x * 256u + threadIdx.x;
  if (t >= (unsigned)E * CG) return;
  const int e  = t / CG;
  const int c4 = (t % CG) * 4;
  const int s = src[e], d = dst[e], r = et[e];
  const float w = ew2[e];
  v4h z4 = *(const v4h*)(zh + ((size_t)r * N + s) * DOUT + c4);
  float* ap = acc + (size_t)d * DOUT + c4;
  atomAddF32(ap + 0, w * (float)z4[0]);
  atomAddF32(ap + 1, w * (float)z4[1]);
  atomAddF32(ap + 2, w * (float)z4[2]);
  atomAddF32(ap + 3, w * (float)z4[3]);
}

// ---------------- bias + relu + f16 cast ----------------
template<int DOUT>
__global__ __launch_bounds__(256) void k_finalize(const float* __restrict__ acc,
                                                  const float* __restrict__ bias,
                                                  _Float16* __restrict__ h, int N) {
  int t = blockIdx.x * 256 + threadIdx.x;
  if (t >= N * DOUT) return;
  int c = t % DOUT;
  h[t] = (_Float16)fmaxf(acc[t] + bias[c], 0.0f);
}

// ---------------- layer 3: z3[v,r] = h2[v] . W3[r]  (d_out = 1) ----------------
__global__ __launch_bounds__(256) void k_z3(const _Float16* __restrict__ h2h,
                                            const float* __restrict__ W3,
                                            float* __restrict__ z3, int N) {
  int t = blockIdx.x * 256 + threadIdx.x;
  if (t >= N * R_REL) return;
  int v = t >> 3, r = t & 7;
  const _Float16* hp = h2h + (size_t)v * 64;
  const float*    wp = W3 + (size_t)r * 64;
  float s = 0.0f;
#pragma unroll 8
  for (int k = 0; k < 64; ++k) s += (float)hp[k] * wp[k];
  z3[t] = s;
}

__global__ __launch_bounds__(256) void k_out_init(float* __restrict__ out,
                                                  const float* __restrict__ b3, int N) {
  int t = blockIdx.x * 256 + threadIdx.x;
  if (t < N) out[t] = b3[0];
}

__global__ __launch_bounds__(256) void k_edge3(const int* __restrict__ src,
                                               const int* __restrict__ dst,
                                               const int* __restrict__ et,
                                               const float* __restrict__ ew2,
                                               const float* __restrict__ z3,
                                               float* __restrict__ out, int E) {
  int t = blockIdx.x * 256 + threadIdx.x;
  if (t >= E) return;
  atomAddF32(&out[dst[t]], ew2[t] * z3[(size_t)src[t] * R_REL + et[t]]);
}

// ---------------- host ----------------
static inline int cdiv(long long a, int b) { return (int)((a + b - 1) / b); }

extern "C" void kernel_launch(void* const* d_in, const int* in_sizes, int n_in,
                              void* d_out, int out_size, void* d_ws, size_t ws_size,
                              hipStream_t stream) {
  const float* b_z  = (const float*)d_in[0];
  const int*   eidx = (const int*)d_in[1];
  const int*   et   = (const int*)d_in[2];
  const float* ew   = (const float*)d_in[3];
  const float* W1   = (const float*)d_in[4];
  const float* b1   = (const float*)d_in[5];
  const float* W2   = (const float*)d_in[6];
  const float* b2   = (const float*)d_in[7];
  const float* W3   = (const float*)d_in[8];
  const float* b3   = (const float*)d_in[9];
  float* out = (float*)d_out;

  const int N = in_sizes[0] / 256;       // 50000
  const int E = in_sizes[2];             // 800000
  const int* src = eidx;
  const int* dst = eidx + E;

  char* ws = (char*)d_ws;
  size_t off = 0;
  auto alloc = [&](size_t bytes) -> void* {
    void* p = ws + off;
    off = (off + bytes + 255) & ~(size_t)255;
    return p;
  };
  float*    cnt  = (float*)   alloc((size_t)N * R_REL * 4);
  float*    ew2  = (float*)   alloc((size_t)E * 4);
  _Float16* xh   = (_Float16*)alloc((size_t)N * 256 * 2);
  _Float16* B1s  = (_Float16*)alloc((size_t)R_REL * 8 * 8 * 32 * 16 * 2);
  _Float16* B2s  = (_Float16*)alloc((size_t)R_REL * 4 * 4 * 32 * 16 * 2);
  _Float16* h1h  = (_Float16*)alloc((size_t)N * 128 * 2);
  _Float16* h2h  = (_Float16*)alloc((size_t)N * 64 * 2);
  float*    z3   = (float*)   alloc((size_t)N * R_REL * 4);
  _Float16* zbig = (_Float16*)alloc((size_t)N * R_REL * 128 * 2);  // z1h; reused as z2h
  float*    abig = (float*)   alloc((size_t)N * 128 * 4);          // acc1; reused as acc2
  (void)ws_size; (void)n_in; (void)out_size;

  // --- graph prep (counts -> folded mean weights) ---
  hipMemsetAsync(cnt, 0, (size_t)N * R_REL * 4, stream);
  k_count<<<cdiv(E, 256), 256, 0, stream>>>(dst, et, cnt, E);
  k_scale_ew<<<cdiv(E, 256), 256, 0, stream>>>(dst, et, ew, cnt, ew2, E);

  // --- conversions / weight swizzle ---
  k_f32_to_f16<<<cdiv((long long)N * 256, 256), 256, 0, stream>>>(b_z, xh, N * 256);
  k_prep_b<256, 128><<<cdiv((long long)R_REL * 8 * 8 * 32 * 16, 256), 256, 0, stream>>>(W1, B1s);
  k_prep_b<128, 64><<<cdiv((long long)R_REL * 4 * 4 * 32 * 16, 256), 256, 0, stream>>>(W2, B2s);

  // --- layer 1: z1 = xh @ W1[r]; scatter; bias+relu ---
  k_gemm<256, 128><<<dim3(cdiv(N, 16), R_REL), 32, 0, stream>>>(xh, B1s, zbig, N);
  hipMemsetAsync(abig, 0, (size_t)N * 128 * 4, stream);
  k_edge_agg<128><<<cdiv((long long)E * 32, 256), 256, 0, stream>>>(src, dst, et, ew2, zbig, abig, E, N);
  k_finalize<128><<<cdiv((long long)N * 128, 256), 256, 0, stream>>>(abig, b1, h1h, N);

  // --- layer 2: z2 = h1 @ W2[r]; scatter; bias+relu ---
  k_gemm<128, 64><<<dim3(cdiv(N, 16), R_REL), 32, 0, stream>>>(h1h, B2s, zbig, N);
  hipMemsetAsync(abig, 0, (size_t)N * 64 * 4, stream);
  k_edge_agg<64><<<cdiv((long long)E * 16, 256), 256, 0, stream>>>(src, dst, et, ew2, zbig, abig, E, N);
  k_finalize<64><<<cdiv((long long)N * 64, 256), 256, 0, stream>>>(abig, b2, h2h, N);

  // --- layer 3: d_out = 1 ---
  k_z3<<<cdiv((long long)N * R_REL, 256), 256, 0, stream>>>(h2h, W3, z3, N);
  k_out_init<<<cdiv(N, 256), 256, 0, stream>>>(out, b3, N);
  k_edge3<<<cdiv(E, 256), 256, 0, stream>>>(src, dst, et, ew2, z3, out, E);
}
